// WaveletnetFull_45277545234506
// MI455X (gfx1250) — compile-verified, matrix operand order checked by reading, not run on hardware
//
#include <hip/hip_runtime.h>
#include <math.h>

typedef __bf16 bf16;
typedef __attribute__((ext_vector_type(16))) __bf16 v16bf;
typedef __attribute__((ext_vector_type(8)))  float  v8f;

union AFrag { v16bf v; bf16 e[16]; };
union CFrag { v8f  v; float e[8]; };

#define WMMA_BF16(A_, B_, C_) \
    __builtin_amdgcn_wmma_f32_16x16x32_bf16(false, (A_), false, (B_), (short)0, (C_), false, false)

__device__ __forceinline__ bf16 f2bf(float f) {
    union { float f; unsigned u; } x; x.f = f;
    unsigned r = (x.u + 0x7FFFu + ((x.u >> 16) & 1u)) >> 16;
    union { unsigned short s; bf16 h; } y; y.s = (unsigned short)r;
    return y.h;
}

// K index inside a 16x32 bf16 A fragment (mirrored for 32x16 B fragment).
// lane L: M (or N) = L&15, kh = L>>4; element i -> VGPR i/2, half i&1.
__device__ __forceinline__ int frag_k(int i, int kh) {
    int v = i >> 1, hb = i & 1;
    int base = (v >= 4) ? 16 : 0;
    return base + 2 * (v & 3) + hb + kh * 8;
}

// ---------------- fragment pack kernels (bandwidth-bound; all divides live here) ----

// A fragments for implicit-GEMM conv3x3 (pad 1): layout [b][stile][kt][lane][16] bf16.
// One thread per [b][stile][kt][lane]; writes one 32B fragment row.
__global__ void im2col_pack_kernel(bf16* __restrict__ ap, const float* __restrict__ in,
                                   int B, int Cin, int H, int W, int Ktiles) {
    const int HW = H * W;
    const int nS = HW >> 4;
    size_t t = (size_t)blockIdx.x * blockDim.x + threadIdx.x;
    size_t total = (size_t)B * nS * Ktiles * 32;
    if (t >= total) return;
    const int lane = (int)(t & 31);
    size_t r = t >> 5;
    const int kt = (int)(r % Ktiles); r /= Ktiles;
    const int stile = (int)(r % nS);
    const int b = (int)(r / nS);
    const int m = lane & 15, kh = lane >> 4;
    const int s = stile * 16 + m;
    const int y = s / W, x = s - y * W;
    const int Ktot = Cin * 9;
    const float* inB = in + (size_t)b * Cin * HW;
    AFrag a;
    #pragma unroll
    for (int i = 0; i < 16; ++i) {
        const int k = kt * 32 + frag_k(i, kh);
        float v = 0.f;
        if (k < Ktot) {
            const int ci = k / 9;
            const int rr = k - ci * 9;
            const int ky = rr / 3, kx = rr - ky * 3;
            const int iy = y + ky - 1, ix = x + kx - 1;
            if (iy >= 0 && iy < H && ix >= 0 && ix < W)
                v = inB[(size_t)ci * HW + (size_t)iy * W + ix];
        }
        a.e[i] = f2bf(v);
    }
    reinterpret_cast<v16bf*>(ap)[t] = a.v;
}

// B fragments: layout [ntile][kt][lane][16] bf16.
// trans=1: src is [N][K] row-major (conv OIHW weights); trans=0: src is [K][N].
__global__ void pack_b_kernel(bf16* __restrict__ bp, const float* __restrict__ src,
                              int K, int N, int Ktiles, int trans) {
    const int Ntiles = (N + 15) >> 4;
    size_t t = (size_t)blockIdx.x * blockDim.x + threadIdx.x;
    size_t total = (size_t)Ntiles * Ktiles * 32;
    if (t >= total) return;
    const int lane = (int)(t & 31);
    size_t r = t >> 5;
    const int kt = (int)(r % Ktiles);
    const int ntile = (int)(r / Ktiles);
    const int n = ntile * 16 + (lane & 15);
    const int kh = lane >> 4;
    AFrag a;
    #pragma unroll
    for (int i = 0; i < 16; ++i) {
        const int k = kt * 32 + frag_k(i, kh);
        float v = 0.f;
        if (k < K && n < N) v = trans ? src[(size_t)n * K + k] : src[(size_t)k * N + n];
        a.e[i] = f2bf(v);
    }
    reinterpret_cast<v16bf*>(bp)[t] = a.v;
}

// A fragments for a single 16-row tile (FC): layout [kt][lane][16] bf16, src [M][K], M<=16.
__global__ void pack_a_rows_kernel(bf16* __restrict__ ap, const float* __restrict__ src,
                                   int M, int K, int Ktiles) {
    size_t t = (size_t)blockIdx.x * blockDim.x + threadIdx.x;
    size_t total = (size_t)Ktiles * 32;
    if (t >= total) return;
    const int lane = (int)(t & 31);
    const int kt = (int)(t >> 5);
    const int m = lane & 15, kh = lane >> 4;
    AFrag a;
    #pragma unroll
    for (int i = 0; i < 16; ++i) {
        const int k = kt * 32 + frag_k(i, kh);
        float v = (k < K && m < M) ? src[(size_t)m * K + k] : 0.f;
        a.e[i] = f2bf(v);
    }
    reinterpret_cast<v16bf*>(ap)[t] = a.v;
}

// ---------------- WMMA GEMM kernels (tight load+wmma loops) ----------------

__device__ __forceinline__ void store_conv_tile(float* __restrict__ out, const v8f& acc,
                                                const float* __restrict__ bias,
                                                int b, int Cout, int HW, int stile,
                                                int ntile, int lane) {
    const int n = lane & 15, kh = lane >> 4;
    const int co = ntile * 16 + n;
    const float bv = bias[co];
    CFrag cf; cf.v = acc;
    float* op = out + ((size_t)b * Cout + co) * HW + stile * 16 + 8 * kh;
    #pragma unroll
    for (int r = 0; r < 8; ++r) op[r] = cf.e[r] + bv;  // HW is a multiple of 16
}

// C[spatial=16, cout=16*NT] per wave; out is NCHW pre-BN activations.
template<int NT>
__global__ __launch_bounds__(128) void gemm_conv_kernel(
    float* __restrict__ out, const bf16* __restrict__ apack, const bf16* __restrict__ bpack,
    const float* __restrict__ bias, int nS, int HW, int Cout, int Ktiles)
{
    const int wave = threadIdx.x >> 5;
    const int lane = threadIdx.x & 31;
    const int stile = blockIdx.x * 4 + wave;
    if (stile >= nS) return;                       // wave-uniform
    const int ng = blockIdx.y;                     // group of NT cout tiles
    const int b = blockIdx.z;
    const v16bf* A  = reinterpret_cast<const v16bf*>(apack)
                    + ((size_t)b * nS + stile) * Ktiles * 32 + lane;
    const v16bf* Bp = reinterpret_cast<const v16bf*>(bpack) + lane;
    const size_t bstr = (size_t)Ktiles * 32;
    size_t aoff = 0;
    size_t boff = (size_t)(ng * NT) * bstr;
    v8f acc0 = {}, acc1 = {}, acc2 = {}, acc3 = {};
    #pragma unroll 2
    for (int kt = 0; kt < Ktiles; ++kt, aoff += 32, boff += 32) {
        const v16bf av = A[aoff];
        acc0 = WMMA_BF16(av, Bp[boff], acc0);
        if constexpr (NT > 1) acc1 = WMMA_BF16(av, Bp[boff + bstr], acc1);
        if constexpr (NT > 2) acc2 = WMMA_BF16(av, Bp[boff + 2 * bstr], acc2);
        if constexpr (NT > 3) acc3 = WMMA_BF16(av, Bp[boff + 3 * bstr], acc3);
    }
    store_conv_tile(out, acc0, bias, b, Cout, HW, stile, ng * NT + 0, lane);
    if constexpr (NT > 1) store_conv_tile(out, acc1, bias, b, Cout, HW, stile, ng * NT + 1, lane);
    if constexpr (NT > 2) store_conv_tile(out, acc2, bias, b, Cout, HW, stile, ng * NT + 2, lane);
    if constexpr (NT > 3) store_conv_tile(out, acc3, bias, b, Cout, HW, stile, ng * NT + 3, lane);
}

__device__ __forceinline__ void store_fc_tile(float* __restrict__ out, const v8f& acc,
                                              const float* __restrict__ bias,
                                              int M, int N, int ntile, int lane, int relu) {
    const int n = lane & 15, kh = lane >> 4;
    const int co = ntile * 16 + n;
    const float bv = bias[co];
    CFrag cf; cf.v = acc;
    #pragma unroll
    for (int r = 0; r < 8; ++r) {
        const int mm = r + 8 * kh;
        if (mm < M) {
            float v = cf.e[r] + bv;
            if (relu) v = fmaxf(v, 0.f);
            out[(size_t)mm * N + co] = v;
        }
    }
}

// out[M,N] = relu?(A[16,K] @ B[K,N] + bias); single M tile, 4 cout tiles per wave.
__global__ __launch_bounds__(128) void gemm_fc_kernel(
    float* __restrict__ out, const bf16* __restrict__ apack, const bf16* __restrict__ bpack,
    const float* __restrict__ bias, int M, int N, int Ktiles, int relu)
{
    const int wave = threadIdx.x >> 5;
    const int lane = threadIdx.x & 31;
    const int ng = blockIdx.x * 4 + wave;
    if (ng * 64 >= N) return;                      // wave-uniform
    const v16bf* A  = reinterpret_cast<const v16bf*>(apack) + lane;
    const v16bf* Bp = reinterpret_cast<const v16bf*>(bpack) + lane;
    const size_t bstr = (size_t)Ktiles * 32;
    size_t aoff = 0;
    size_t boff = (size_t)(ng * 4) * bstr;
    v8f acc0 = {}, acc1 = {}, acc2 = {}, acc3 = {};
    #pragma unroll 2
    for (int kt = 0; kt < Ktiles; ++kt, aoff += 32, boff += 32) {
        const v16bf av = A[aoff];
        acc0 = WMMA_BF16(av, Bp[boff], acc0);
        acc1 = WMMA_BF16(av, Bp[boff + bstr], acc1);
        acc2 = WMMA_BF16(av, Bp[boff + 2 * bstr], acc2);
        acc3 = WMMA_BF16(av, Bp[boff + 3 * bstr], acc3);
    }
    store_fc_tile(out, acc0, bias, M, N, ng * 4 + 0, lane, relu);
    store_fc_tile(out, acc1, bias, M, N, ng * 4 + 1, lane, relu);
    store_fc_tile(out, acc2, bias, M, N, ng * 4 + 2, lane, relu);
    store_fc_tile(out, acc3, bias, M, N, ng * 4 + 3, lane, relu);
}

// ---------------- wavelet / elementwise kernels ----------------

__global__ void wt_kernel(float* __restrict__ out, const float* __restrict__ in,
                          int B, int C, int H, int W) {
    int h = H >> 1, w = W >> 1;
    size_t t = (size_t)blockIdx.x * blockDim.x + threadIdx.x;
    size_t total = (size_t)B * C * h * w;
    if (t >= total) return;
    int xw = (int)(t % w); size_t r = t / w;
    int yh = (int)(r % h); r /= h;
    int c  = (int)(r % C); int b = (int)(r / C);
    const float* p = in + ((size_t)(b * C + c) * H + 2 * yh) * W + 2 * xw;
    float x00 = p[0], x01 = p[1], x10 = p[W], x11 = p[W + 1];
    float b0 = 0.25f * (x00 + x01 + x10 + x11);
    float b1 = 0.5f  * (x00 + x01 - x10 - x11);
    float b2 = 0.5f  * (x00 - x01 + x10 - x11);
    float b3 = 0.5f  * (x00 - x01 - x10 + x11);
    b1 = (b1 + 1.f) * 0.5f; b2 = (b2 + 1.f) * 0.5f; b3 = (b3 + 1.f) * 0.5f;
    size_t hw = (size_t)h * w;
    float* q = out + ((size_t)b * 4 * C + (size_t)c * 4) * hw + (size_t)yh * w + xw;
    q[0] = b0; q[hw] = b1; q[2 * hw] = b2; q[3 * hw] = b3;
}

__global__ void iwt_kernel(float* __restrict__ out, const float* __restrict__ in,
                           int B, int C4, int h, int w) {
    int C = C4 >> 2;
    size_t t = (size_t)blockIdx.x * blockDim.x + threadIdx.x;
    size_t total = (size_t)B * C * h * w;
    if (t >= total) return;
    int xw = (int)(t % w); size_t r = t / w;
    int yh = (int)(r % h); r /= h;
    int c  = (int)(r % C); int b = (int)(r / C);
    size_t hw = (size_t)h * w;
    const float* p = in + ((size_t)b * C4 + (size_t)c * 4) * hw + (size_t)yh * w + xw;
    float v0 = p[0];
    float v1 = 2.f * p[hw]     - 1.f;
    float v2 = 2.f * p[2 * hw] - 1.f;
    float v3 = 2.f * p[3 * hw] - 1.f;
    int W2 = 2 * w;
    float* q = out + ((size_t)b * C + c) * (4 * hw) + (size_t)(2 * yh) * W2 + 2 * xw;
    q[0]      = v0 + 0.5f * ( v1 + v2 + v3);
    q[1]      = v0 + 0.5f * ( v1 - v2 - v3);
    q[W2]     = v0 + 0.5f * (-v1 + v2 - v3);
    q[W2 + 1] = v0 + 0.5f * (-v1 - v2 + v3);
}

__global__ void copy_chan_kernel(float* __restrict__ dst, const float* __restrict__ src,
                                 int B, int Cd, int c0, int Cs, int HW) {
    size_t t = (size_t)blockIdx.x * blockDim.x + threadIdx.x;
    size_t n = (size_t)B * Cs * HW;
    if (t >= n) return;
    int s = (int)(t % HW); size_t r = t / HW;
    int c = (int)(r % Cs); int b = (int)(r / Cs);
    dst[((size_t)b * Cd + c0 + c) * HW + s] = src[t];
}

// ---------------- BatchNorm ----------------

__global__ void bn_stats_kernel(float* __restrict__ stats, const float* __restrict__ x,
                                int B, int C, int HW) {
    int c = blockIdx.x;
    __shared__ float ssum[256], ssq[256];
    float s = 0.f, q = 0.f;
    for (int b = 0; b < B; ++b) {
        const float* p = x + ((size_t)b * C + c) * HW;
        for (int i = threadIdx.x; i < HW; i += blockDim.x) {
            float v = p[i]; s += v; q += v * v;
        }
    }
    ssum[threadIdx.x] = s; ssq[threadIdx.x] = q;
    __syncthreads();
    for (int st = 128; st > 0; st >>= 1) {
        if ((int)threadIdx.x < st) {
            ssum[threadIdx.x] += ssum[threadIdx.x + st];
            ssq[threadIdx.x]  += ssq[threadIdx.x + st];
        }
        __syncthreads();
    }
    if (threadIdx.x == 0) {
        float n = (float)B * (float)HW;
        float mean = ssum[0] / n;
        float var  = ssq[0] / n - mean * mean;
        stats[2 * c] = mean; stats[2 * c + 1] = var;
    }
}

__global__ void bn_relu_kernel(float* __restrict__ out, const float* __restrict__ x,
                               const float* __restrict__ stats,
                               const float* __restrict__ g, const float* __restrict__ be,
                               int C, int HW, size_t n) {
    size_t t = (size_t)blockIdx.x * blockDim.x + threadIdx.x;
    if (t >= n) return;
    int c = (int)((t / HW) % C);
    float v = g[c] * (x[t] - stats[2 * c]) * rsqrtf(stats[2 * c + 1] + 1e-5f) + be[c];
    out[t] = fmaxf(v, 0.f);
}

// ---------------- heads ----------------

__global__ void outc_kernel(float* __restrict__ out, const float* __restrict__ u,
                            const float* __restrict__ w, const float* __restrict__ bias,
                            int B, int Cin, int Cout, int HW) {
    size_t t = (size_t)blockIdx.x * blockDim.x + threadIdx.x;
    size_t n = (size_t)B * Cout * HW;
    if (t >= n) return;
    int s  = (int)(t % HW); size_t r = t / HW;
    int co = (int)(r % Cout); int b = (int)(r / Cout);
    float acc = bias[co];
    const float* up = u + (size_t)b * Cin * HW + s;
    for (int ci = 0; ci < Cin; ++ci) acc += w[co * Cin + ci] * up[(size_t)ci * HW];
    out[t] = acc;
}

__global__ void sigmoid_kernel(float* __restrict__ x, size_t n) {
    size_t t = (size_t)blockIdx.x * blockDim.x + threadIdx.x;
    if (t < n) x[t] = 1.f / (1.f + expf(-x[t]));
}

__global__ void gap_kernel(float* __restrict__ feat, const float* __restrict__ x,
                           int B, int C, int HW) {
    int t = blockIdx.x * blockDim.x + threadIdx.x;
    if (t >= B * C) return;
    const float* p = x + (size_t)t * HW;
    float s = 0.f;
    for (int i = 0; i < HW; ++i) s += p[i];
    feat[t] = s / (float)HW;
}

__global__ void head_kernel(float* __restrict__ out, const float* __restrict__ h,
                            const float* __restrict__ w2, const float* __restrict__ b2, int K) {
    int b = blockIdx.x;
    __shared__ float red[256];
    float s = 0.f;
    for (int k = threadIdx.x; k < K; k += blockDim.x) s += h[(size_t)b * K + k] * w2[k];
    red[threadIdx.x] = s; __syncthreads();
    for (int st = 128; st > 0; st >>= 1) {
        if ((int)threadIdx.x < st) red[threadIdx.x] += red[threadIdx.x + st];
        __syncthreads();
    }
    if (threadIdx.x == 0) out[b] = red[0] + b2[0];
}

// ---------------- host ----------------

static inline size_t cdiv(size_t a, size_t b) { return (a + b - 1) / b; }

static void launch_conv(hipStream_t st, const float* in, int B, int Cin, int H, int W, int Cout,
                        const float* w, const float* bias, const float* g, const float* be,
                        float* out, float* pre, bf16* apack, bf16* bpack, float* stats)
{
    const int HW = H * W;
    const int nS = HW >> 4;
    const int Ktot = Cin * 9;
    const int Ktiles = (Ktot + 31) / 32;
    const int CoutTiles = (Cout + 15) / 16;
    size_t na = (size_t)B * nS * Ktiles * 32;
    im2col_pack_kernel<<<(unsigned)cdiv(na, 256), 256, 0, st>>>(apack, in, B, Cin, H, W, Ktiles);
    size_t nb = (size_t)CoutTiles * Ktiles * 32;
    pack_b_kernel<<<(unsigned)cdiv(nb, 256), 256, 0, st>>>(bpack, w, Ktot, Cout, Ktiles, 1);
    if (CoutTiles % 4 == 0) {
        dim3 grid((unsigned)cdiv(nS, 4), (unsigned)(CoutTiles / 4), (unsigned)B);
        gemm_conv_kernel<4><<<grid, 128, 0, st>>>(pre, apack, bpack, bias, nS, HW, Cout, Ktiles);
    } else {
        dim3 grid((unsigned)cdiv(nS, 4), (unsigned)CoutTiles, (unsigned)B);
        gemm_conv_kernel<1><<<grid, 128, 0, st>>>(pre, apack, bpack, bias, nS, HW, Cout, Ktiles);
    }
    bn_stats_kernel<<<Cout, 256, 0, st>>>(stats, pre, B, Cout, HW);
    size_t nout = (size_t)B * Cout * HW;
    bn_relu_kernel<<<(unsigned)cdiv(nout, 256), 256, 0, st>>>(out, pre, stats, g, be, Cout, HW, nout);
}

static void double_conv(hipStream_t st, const float* in, int B, int Cin, int H, int W, int Cout,
                        void* const* d_in, int base, float* out, float* mid, float* pre,
                        bf16* apack, bf16* bpack, float* stats)
{
    launch_conv(st, in, B, Cin, H, W, Cout,
                (const float*)d_in[base + 0], (const float*)d_in[base + 1],
                (const float*)d_in[base + 2], (const float*)d_in[base + 3],
                mid, pre, apack, bpack, stats);
    launch_conv(st, mid, B, Cout, H, W, Cout,
                (const float*)d_in[base + 4], (const float*)d_in[base + 5],
                (const float*)d_in[base + 6], (const float*)d_in[base + 7],
                out, pre, apack, bpack, stats);
}

extern "C" void kernel_launch(void* const* d_in, const int* in_sizes, int n_in,
                              void* d_out, int out_size, void* d_ws, size_t ws_size,
                              hipStream_t stream) {
    (void)in_sizes; (void)n_in; (void)out_size; (void)ws_size;
    const int B = 8;
    const float* X = (const float*)d_in[0];
    float* dout = (float*)d_out;

    char* ws = (char*)d_ws;
    size_t off = 0;
    auto alloc = [&](size_t bytes) -> void* {
        void* p = ws + off;
        off = (off + bytes + 255) & ~(size_t)255;
        return p;
    };
    const size_t F8M = 8388608;
    float* wtbuf  = (float*)alloc(F8M * 4);
    float* c1     = (float*)alloc(F8M * 4);
    float* c2     = (float*)alloc(F8M * 4);
    float* c3     = (float*)alloc(F8M * 4);
    float* c4     = (float*)alloc(F8M * 4);
    float* mid    = (float*)alloc(F8M * 4);
    float* pre    = (float*)alloc(F8M * 4);
    float* iwtbuf = (float*)alloc(F8M * 4);
    float* ubuf   = (float*)alloc(F8M * 4);
    float* catbuf = (float*)alloc(2 * F8M * 4);
    float* outcb  = (float*)alloc((size_t)2097152 * 4);
    float* feat   = (float*)alloc(8192 * 4);
    float* hbuf   = (float*)alloc(8192 * 4);
    float* stats  = (float*)alloc(2048 * 4);
    bf16*  apack  = (bf16*)alloc((size_t)150994944 * 2);  // max im2col fragment buffer
    bf16*  bpack  = (bf16*)alloc((size_t)9437184 * 2);

    // ---- encoder ----
    wt_kernel<<<(unsigned)cdiv((size_t)B * 1 * 256 * 256, 256), 256, 0, stream>>>(wtbuf, X, B, 1, 512, 512);
    double_conv(stream, wtbuf, B, 4, 256, 256, 16, d_in, 1, c1, mid, pre, apack, bpack, stats);

    wt_kernel<<<(unsigned)cdiv((size_t)B * 16 * 128 * 128, 256), 256, 0, stream>>>(wtbuf, c1, B, 16, 256, 256);
    double_conv(stream, wtbuf, B, 64, 128, 128, 64, d_in, 9, c2, mid, pre, apack, bpack, stats);

    wt_kernel<<<(unsigned)cdiv((size_t)B * 64 * 64 * 64, 256), 256, 0, stream>>>(wtbuf, c2, B, 64, 128, 128);
    double_conv(stream, wtbuf, B, 256, 64, 64, 256, d_in, 17, c3, mid, pre, apack, bpack, stats);

    wt_kernel<<<(unsigned)cdiv((size_t)B * 256 * 32 * 32, 256), 256, 0, stream>>>(wtbuf, c3, B, 256, 64, 64);
    double_conv(stream, wtbuf, B, 1024, 32, 32, 1024, d_in, 25, c4, mid, pre, apack, bpack, stats);

    // ---- decoder ----
    iwt_kernel<<<(unsigned)cdiv((size_t)B * 256 * 32 * 32, 256), 256, 0, stream>>>(iwtbuf, c4, B, 1024, 32, 32);
    copy_chan_kernel<<<(unsigned)cdiv((size_t)B * 256 * 4096, 256), 256, 0, stream>>>(catbuf, c3, B, 512, 0, 256, 4096);
    copy_chan_kernel<<<(unsigned)cdiv((size_t)B * 256 * 4096, 256), 256, 0, stream>>>(catbuf, iwtbuf, B, 512, 256, 256, 4096);
    double_conv(stream, catbuf, B, 512, 64, 64, 256, d_in, 33, ubuf, mid, pre, apack, bpack, stats);

    iwt_kernel<<<(unsigned)cdiv((size_t)B * 64 * 64 * 64, 256), 256, 0, stream>>>(iwtbuf, ubuf, B, 256, 64, 64);
    copy_chan_kernel<<<(unsigned)cdiv((size_t)B * 64 * 16384, 256), 256, 0, stream>>>(catbuf, c2, B, 128, 0, 64, 16384);
    copy_chan_kernel<<<(unsigned)cdiv((size_t)B * 64 * 16384, 256), 256, 0, stream>>>(catbuf, iwtbuf, B, 128, 64, 64, 16384);
    double_conv(stream, catbuf, B, 128, 128, 128, 64, d_in, 41, ubuf, mid, pre, apack, bpack, stats);

    iwt_kernel<<<(unsigned)cdiv((size_t)B * 16 * 128 * 128, 256), 256, 0, stream>>>(iwtbuf, ubuf, B, 64, 128, 128);
    copy_chan_kernel<<<(unsigned)cdiv((size_t)B * 16 * 65536, 256), 256, 0, stream>>>(catbuf, c1, B, 32, 0, 16, 65536);
    copy_chan_kernel<<<(unsigned)cdiv((size_t)B * 16 * 65536, 256), 256, 0, stream>>>(catbuf, iwtbuf, B, 32, 16, 16, 65536);
    double_conv(stream, catbuf, B, 32, 256, 256, 16, d_in, 49, ubuf, mid, pre, apack, bpack, stats);

    // ---- output head ----
    outc_kernel<<<(unsigned)cdiv((size_t)B * 4 * 65536, 256), 256, 0, stream>>>(
        outcb, ubuf, (const float*)d_in[57], (const float*)d_in[58], B, 16, 4, 65536);
    iwt_kernel<<<(unsigned)cdiv((size_t)B * 1 * 65536, 256), 256, 0, stream>>>(dout, outcb, B, 4, 256, 256);
    sigmoid_kernel<<<(unsigned)cdiv((size_t)2097152, 256), 256, 0, stream>>>(dout, (size_t)2097152);

    // ---- confidence scorer ----
    gap_kernel<<<32, 256, 0, stream>>>(feat, c4, B, 1024, 1024);
    {
        const int K = 1024, N = 1024, Ktiles = 32;
        pack_a_rows_kernel<<<(unsigned)cdiv((size_t)Ktiles * 32, 256), 256, 0, stream>>>(apack, feat, B, K, Ktiles);
        pack_b_kernel<<<(unsigned)cdiv((size_t)64 * Ktiles * 32, 256), 256, 0, stream>>>(
            bpack, (const float*)d_in[59], K, N, Ktiles, 0);
        gemm_fc_kernel<<<4, 128, 0, stream>>>(hbuf, apack, bpack, (const float*)d_in[60], B, N, Ktiles, 1);
    }
    head_kernel<<<8, 256, 0, stream>>>(dout + 2097152, hbuf, (const float*)d_in[61],
                                       (const float*)d_in[62], 1024);
}
